// WaveNetEncoder_71708773974274
// MI455X (gfx1250) — matrix operand order repaired and checked by reading, not compiled, not run on hardware
//
#include <hip/hip_runtime.h>

// ---------------------------------------------------------------------------
// WaveNet encoder, fully fused in LDS, f16 WMMA on gfx1250 (CDNA5, wave32).
//
// Each block owns one 256-sample pooling window of one batch element.
// Receptive field = sum(dilations) = 510, so the block loads a 766-sample
// signal window, runs all 16 residual layers inside LDS (ping-pong f16
// buffers), and emits a single pooled float. All channel-mixing matmuls use
// V_WMMA_F32_16X16X32_F16 (M = out-channels, K = in-channels, N = time).
// ReLU = packed signed-int max (v_pk_max_i16); f32->f16 = v_cvt_pk_rtz.
// ---------------------------------------------------------------------------

typedef __attribute__((ext_vector_type(16))) _Float16     v16h;
typedef __attribute__((ext_vector_type(8)))  _Float16     v8h;
typedef __attribute__((ext_vector_type(16))) short        v16s;
typedef __attribute__((ext_vector_type(8)))  short        v8s;
typedef __attribute__((ext_vector_type(8)))  float        v8f;
typedef __attribute__((ext_vector_type(8)))  unsigned int v8u;

#define TPB 256

constexpr int T_IN   = 131072;
constexpr int TILE_T = 256;          // pre-pool samples per block (= 1 pool win)
constexpr int HALO   = 510;          // sum of dilations over 16 layers
constexpr int LEN0   = TILE_T + HALO; // 766
constexpr int XROWS  = 784;          // 766 + tile-overrun slack
constexpr int XPITCH = 72;           // f16 per time-row: 64 ch + pad (144 B)
constexpr int HPITCH = 136;          // f16 per h column: 128 ch + pad (272 B)
constexpr int NPOOL  = 510;
constexpr int CR = 64, CD = 128;

// ------------------------- LDS layout (byte offsets) -----------------------
constexpr size_t XBYTES  = (size_t)XROWS * XPITCH * 2;          // 112896
constexpr size_t OFF_X0  = 0;
constexpr size_t OFF_X1  = XBYTES;
constexpr size_t OFF_WD0 = 2 * XBYTES;
constexpr size_t OFF_WD1 = OFF_WD0 + (size_t)CD * CR * 2;
constexpr size_t OFF_WR  = OFF_WD1 + (size_t)CD * CR * 2;
constexpr size_t OFF_BD  = OFF_WR  + (size_t)CR * CD * 2;       // 128 f32 bias
constexpr size_t OFF_BR  = OFF_BD + 128 * 4;                    // 64 f32 bias
constexpr size_t OFF_HS  = OFF_BR + 64 * 4;                     // per-wave h scratch
constexpr size_t SMEM_BYTES = OFF_HS + (size_t)8 * 16 * HPITCH * 2; // 310528
static_assert(SMEM_BYTES <= 320u * 1024u, "LDS budget (320KB/WGP)");

// ------------------------------ helpers ------------------------------------
// f32 pair -> packed f16 (single v_cvt_pk_rtz_f16_f32)
static __device__ __forceinline__ unsigned pk2(float a, float b) {
  auto h = __builtin_amdgcn_cvt_pkrtz(a, b);
  return __builtin_bit_cast(unsigned, h);
}
static __device__ __forceinline__ v16s z16() {
  v16s z;
#pragma unroll
  for (int i = 0; i < 16; ++i) z[i] = 0;
  return z;
}
static __device__ __forceinline__ v8s z8() {
  v8s z;
#pragma unroll
  for (int i = 0; i < 8; ++i) z[i] = 0;
  return z;
}
// packed relu on 8 f16 held in a uint4 (sign-magnitude => signed-int max)
static __device__ __forceinline__ uint4 relu_pk4(uint4 v) {
  v8s s = __builtin_bit_cast(v8s, v);
  s = __builtin_elementwise_max(s, z8());
  return __builtin_bit_cast(uint4, s);
}
// B fragment: 16 consecutive f16 from LDS (lane = column N, K contiguous)
static __device__ __forceinline__ v16h ld16(const unsigned short* p) {
  const uint4* q = (const uint4*)p;
  uint4 a = q[0], b = q[1];
  v8u r; r[0]=a.x; r[1]=a.y; r[2]=a.z; r[3]=a.w; r[4]=b.x; r[5]=b.y; r[6]=b.z; r[7]=b.w;
  return __builtin_bit_cast(v16h, r);
}
// same, with fused relu (8x v_pk_max_i16)
static __device__ __forceinline__ v16h ld16_relu(const unsigned short* p) {
  const uint4* q = (const uint4*)p;
  uint4 a = q[0], b = q[1];
  v8u r; r[0]=a.x; r[1]=a.y; r[2]=a.z; r[3]=a.w; r[4]=b.x; r[5]=b.y; r[6]=b.z; r[7]=b.w;
  v16s s = __builtin_bit_cast(v16s, r);
  s = __builtin_elementwise_max(s, z16());
  return __builtin_bit_cast(v16h, s);
}
// A fragment: lane holds row M, K contiguous [kbase..kbase+15] in v0..v7.
static __device__ __forceinline__ v16h ldA(const unsigned short* p) {
  uint4 a = *(const uint4*)p;
  uint4 b = *(const uint4*)(p + 8);
  v8u r; r[0]=a.x; r[1]=a.y; r[2]=a.z; r[3]=a.w; r[4]=b.x; r[5]=b.y; r[6]=b.z; r[7]=b.w;
  return __builtin_bit_cast(v16h, r);
}
// 8 consecutive f32 from LDS (bias -> accumulator init, D-layout rows)
static __device__ __forceinline__ v8f ldb8(const float* p) {
  float4 a = *(const float4*)p;
  float4 b = *(const float4*)(p + 4);
  v8f r; r[0]=a.x; r[1]=a.y; r[2]=a.z; r[3]=a.w; r[4]=b.x; r[5]=b.y; r[6]=b.z; r[7]=b.w;
  return r;
}

// ------------------------------- kernel ------------------------------------
__global__ __launch_bounds__(TPB, 1)
void wavenet_enc_fused(const float* __restrict__ sig,
                       const float* __restrict__ W0,
                       const float* __restrict__ b0,
                       const float* __restrict__ Wd,
                       const float* __restrict__ bd,
                       const float* __restrict__ Wr,
                       const float* __restrict__ br,
                       const float* __restrict__ Wf,
                       const float* __restrict__ bfp,
                       float* __restrict__ out)
{
  extern __shared__ char smem[];
  unsigned short* X0  = (unsigned short*)(smem + OFF_X0);
  unsigned short* X1  = (unsigned short*)(smem + OFF_X1);
  unsigned short* WD0 = (unsigned short*)(smem + OFF_WD0);
  unsigned short* WD1 = (unsigned short*)(smem + OFF_WD1);
  unsigned short* WRS = (unsigned short*)(smem + OFF_WR);
  float*          BDs = (float*)(smem + OFF_BD);
  float*          BRs = (float*)(smem + OFF_BR);
  unsigned short* HS  = (unsigned short*)(smem + OFF_HS);

  const int tid  = threadIdx.x;
  const int lane = tid & 31;
  const int wave = tid >> 5;
  const int lcol = lane & 15;        // column N within 16x16 tile
  const int lhalf = lane >> 4;       // 0 or 1 (K / M half selector)
  const int chb  = lhalf * 8;        // row sub-offset inside a 16-row M tile

  const int j = blockIdx.x;          // pooled output index [0,510)
  const int b = blockIdx.y;          // batch

  // ---- stage 0: input 1x1 conv signal -> x0[64 x LEN0], f16, time-major ---
  if (tid < CR) { BDs[tid] = W0[tid]; BRs[tid] = b0[tid]; }
  __syncthreads();

  const float* sg = sig + (size_t)b * T_IN + (size_t)j * TILE_T;
  for (int t = tid; t < LEN0; t += TPB) {
    float s = sg[t];
    unsigned short* row = X0 + (size_t)t * XPITCH;
#pragma unroll
    for (int c0 = 0; c0 < CR; c0 += 8) {
      uint4 v;
      v.x = pk2(BDs[c0+0]*s + BRs[c0+0], BDs[c0+1]*s + BRs[c0+1]);
      v.y = pk2(BDs[c0+2]*s + BRs[c0+2], BDs[c0+3]*s + BRs[c0+3]);
      v.z = pk2(BDs[c0+4]*s + BRs[c0+4], BDs[c0+5]*s + BRs[c0+5]);
      v.w = pk2(BDs[c0+6]*s + BRs[c0+6], BDs[c0+7]*s + BRs[c0+7]);
      *(uint4*)(row + c0) = v;
    }
  }
  // deterministic zero padding rows (tile overrun region) in both buffers
  for (int t = LEN0 + tid; t < XROWS; t += TPB) {
    unsigned short* r0 = X0 + (size_t)t * XPITCH;
    unsigned short* r1 = X1 + (size_t)t * XPITCH;
    uint4 z; z.x = z.y = z.z = z.w = 0u;
#pragma unroll
    for (int c0 = 0; c0 < XPITCH; c0 += 8) { *(uint4*)(r0 + c0) = z; *(uint4*)(r1 + c0) = z; }
  }
  __syncthreads();

  // --------------------------- 16 residual layers --------------------------
  unsigned short* xin  = X0;
  unsigned short* xout = X1;
  int len = LEN0;

  for (int li = 0; li < 16; ++li) {
    const int d = 1 << (li & 7);

    // stage this layer's weights into LDS as f16 (vector loads + pkrtz)
    const float* wdG = Wd + (size_t)li * CD * CR * 2;   // [o][c][k]
    for (int i = tid * 2; i < CD * CR; i += TPB * 2) {
      float4 w4 = *(const float4*)(wdG + 2 * i);        // (i,k0)(i,k1)(i+1,k0)(i+1,k1)
      *(unsigned*)(WD0 + i) = pk2(w4.x, w4.z);
      *(unsigned*)(WD1 + i) = pk2(w4.y, w4.w);
    }
    const float* wrG = Wr + (size_t)li * CR * CD;       // [o][c]
    for (int i = tid * 2; i < CR * CD; i += TPB * 2) {
      float2 w2 = *(const float2*)(wrG + i);
      *(unsigned*)(WRS + i) = pk2(w2.x, w2.y);
    }
    if (tid < CD) BDs[tid] = bd[(size_t)li * CD + tid];
    if (tid < CR) BRs[tid] = br[(size_t)li * CR + tid];
    if (li < 15 && tid == 0)
      __builtin_prefetch(Wd + (size_t)(li + 1) * CD * CR * 2, 0, 0); // global_prefetch_b8
    __syncthreads();

    const int Lout = len - d;
    const int ntiles = (Lout + 15) >> 4;
    unsigned short* hs = HS + (size_t)wave * 16 * HPITCH;

    for (int tile = wave; tile < ntiles; tile += 8) {
      const int t = tile << 4;

      // ---- dilated conv: h[128 x 16] = Wd0*relu(x[t..]) + Wd1*relu(x[t+d..])
      // accumulators initialized with the bias (row bias broadcast over cols)
      v8f acc[8];
#pragma unroll
      for (int m = 0; m < 8; ++m) acc[m] = ldb8(BDs + m * 16 + chb);

#pragma unroll
      for (int k0 = 0; k0 < CR; k0 += 32) {
        const int kb = k0 + lhalf * 16;
        v16h B0 = ld16_relu(xin + (size_t)(t + lcol    ) * XPITCH + kb);
        v16h B1 = ld16_relu(xin + (size_t)(t + lcol + d) * XPITCH + kb);
#pragma unroll
        for (int m = 0; m < 8; ++m) {
          v16h A0 = ldA(WD0 + (size_t)(m * 16 + lcol) * CR + k0 + lhalf * 16);
          acc[m] = __builtin_amdgcn_wmma_f32_16x16x32_f16(
                       false, A0, false, B0, (short)0, acc[m], false, false);
          v16h A1 = ldA(WD1 + (size_t)(m * 16 + lcol) * CR + k0 + lhalf * 16);
          acc[m] = __builtin_amdgcn_wmma_f32_16x16x32_f16(
                       false, A1, false, B1, (short)0, acc[m], false, false);
        }
      }

      // pack to f16 (v_cvt_pk_rtz) + packed relu (v_pk_max_i16), store h
#pragma unroll
      for (int m = 0; m < 8; ++m) {
        uint4 v;
        v.x = pk2(acc[m][0], acc[m][1]);
        v.y = pk2(acc[m][2], acc[m][3]);
        v.z = pk2(acc[m][4], acc[m][5]);
        v.w = pk2(acc[m][6], acc[m][7]);
        v = relu_pk4(v);
        *(uint4*)(hs + (size_t)lcol * HPITCH + m * 16 + chb) = v;
      }

      // ---- residual 1x1 conv: r[64 x 16] = Wr * h  (K = 128), bias in acc
      v8f racc[4];
#pragma unroll
      for (int m = 0; m < 4; ++m) racc[m] = ldb8(BRs + m * 16 + chb);

#pragma unroll
      for (int k0 = 0; k0 < CD; k0 += 32) {
        v16h Bh = ld16(hs + (size_t)lcol * HPITCH + k0 + lhalf * 16);
#pragma unroll
        for (int m = 0; m < 4; ++m) {
          v16h A = ldA(WRS + (size_t)(m * 16 + lcol) * CD + k0 + lhalf * 16);
          racc[m] = __builtin_amdgcn_wmma_f32_16x16x32_f16(
                        false, A, false, Bh, (short)0, racc[m], false, false);
        }
      }

      // skip add (right-aligned x[t+d]) done packed in f16 (v_pk_add_f16)
#pragma unroll
      for (int m = 0; m < 4; ++m) {
        uint4 s4 = *(const uint4*)(xin + (size_t)(t + lcol + d) * XPITCH + m * 16 + chb);
        uint4 pr;
        pr.x = pk2(racc[m][0], racc[m][1]);
        pr.y = pk2(racc[m][2], racc[m][3]);
        pr.z = pk2(racc[m][4], racc[m][5]);
        pr.w = pk2(racc[m][6], racc[m][7]);
        v8h o = __builtin_bit_cast(v8h, pr) + __builtin_bit_cast(v8h, s4);
        *(uint4*)(xout + (size_t)(t + lcol) * XPITCH + m * 16 + chb) =
            __builtin_bit_cast(uint4, o);
      }
    } // tiles

    __syncthreads();
    unsigned short* tmp = xin; xin = xout; xout = tmp;
    len = Lout;
  } // layers

  // ---- epilogue: final 1x1 conv (Wf) + 256-wide average pool --------------
  // pooled = sum_c Wf[c] * (sum_t x[c,t]) / 256 + bf
  float* red = (float*)HS; // reuse scratch (safe after barrier)
  {
    const _Float16* xh = (const _Float16*)xin;
    const int c  = tid & 63;
    const int tq = tid >> 6; // 0..3
    float s = 0.0f;
    for (int t = tq; t < TILE_T; t += 4)
      s += (float)xh[(size_t)t * XPITCH + c];
    s *= Wf[c];
    red[tid] = s;
  }
  __syncthreads();
#pragma unroll
  for (int off = TPB / 2; off > 0; off >>= 1) {
    if (tid < off) red[tid] += red[tid + off];
    __syncthreads();
  }
  if (tid == 0)
    out[(size_t)b * NPOOL + j] = red[0] * (1.0f / 256.0f) + bfp[0];
}

// ------------------------------ launcher -----------------------------------
extern "C" void kernel_launch(void* const* d_in, const int* in_sizes, int n_in,
                              void* d_out, int out_size, void* d_ws, size_t ws_size,
                              hipStream_t stream) {
  (void)in_sizes; (void)n_in; (void)out_size; (void)d_ws; (void)ws_size;
  const float* sig = (const float*)d_in[0];
  const float* W0  = (const float*)d_in[1];
  const float* b0  = (const float*)d_in[2];
  const float* Wd  = (const float*)d_in[3];
  const float* bd  = (const float*)d_in[4];
  const float* Wr  = (const float*)d_in[5];
  const float* br  = (const float*)d_in[6];
  const float* Wf  = (const float*)d_in[7];
  const float* bfp = (const float*)d_in[8];
  float* out = (float*)d_out;

  (void)hipFuncSetAttribute((const void*)wavenet_enc_fused,
                            hipFuncAttributeMaxDynamicSharedMemorySize,
                            (int)SMEM_BYTES);

  dim3 grid(NPOOL, 8);
  wavenet_enc_fused<<<grid, TPB, SMEM_BYTES, stream>>>(
      sig, W0, b0, Wd, bd, Wr, br, Wf, bfp, out);
}
